// MHA_45732811768212
// MI455X (gfx1250) — matrix operand order repaired and checked by reading, not compiled, not run on hardware
//
#include <hip/hip_runtime.h>
#include <hip/hip_bf16.h>

// ---------------------------------------------------------------------------
// GQA attention block for MI455X (gfx1250, wave32, WMMA bf16->f32,
// async global->LDS staging, double-buffered GEMM tiles)
// ---------------------------------------------------------------------------

typedef __bf16 bf16_t;
typedef __attribute__((ext_vector_type(16))) __bf16 v16bf;
typedef __attribute__((ext_vector_type(8)))  float  v8f;

#define B_   2
#define S_   2048
#define D_   2048
#define H_   32
#define DH_  64
#define G_   2
#define KV_  128          // G_ * DH_
#define EPS_ 1e-8f

static __device__ __forceinline__ v8f wmma_bf16(v16bf a, v16bf b, v8f c) {
  return __builtin_amdgcn_wmma_f32_16x16x32_bf16(
      false, a, false, b, (short)0, c, false, false);
}

// CDNA5 async copy: 16B global -> LDS, tracked by ASYNCcnt (no VGPR staging).
// VDST operand is the LDS byte address (low 32 bits of the generic pointer).
static __device__ __forceinline__ void async_copy_b128(void* lds, const void* g) {
  unsigned l = (unsigned)(size_t)lds;
  asm volatile("global_load_async_to_lds_b128 %0, %1, off"
               :: "v"(l), "v"(g) : "memory");
}
static __device__ __forceinline__ void wait_async0() {
  asm volatile("s_wait_asynccnt 0x0" ::: "memory");
}

// ---------------------------------------------------------------------------
// 1) RMS_split pre-norm: h = norm_scale * x / (||x||/sqrt(d) + eps) * split
// ---------------------------------------------------------------------------
__global__ __launch_bounds__(256)
void rmsnorm_kernel(const float* __restrict__ x,
                    const float* __restrict__ nscale,
                    const float* __restrict__ sscale,
                    bf16_t* __restrict__ h) {
  __shared__ float red[256];
  const int row = blockIdx.x;
  const float* xr = x + (size_t)row * D_;
  float ss = 0.f;
  for (int i = threadIdx.x; i < D_; i += 256) { float v = xr[i]; ss += v * v; }
  red[threadIdx.x] = ss;
  __syncthreads();
  for (int off = 128; off > 0; off >>= 1) {
    if (threadIdx.x < off) red[threadIdx.x] += red[threadIdx.x + off];
    __syncthreads();
  }
  const float nrm = sqrtf(red[0] * (1.0f / (float)D_));
  const float inv = 1.0f / (nrm + EPS_);
  bf16_t* hr = h + (size_t)row * D_;
  for (int i = threadIdx.x; i < D_; i += 256)
    hr[i] = (bf16_t)(nscale[i] * xr[i] * inv * sscale[i]);
}

// ---------------------------------------------------------------------------
// 2) f32 -> bf16 conversion (weights)
// ---------------------------------------------------------------------------
__global__ __launch_bounds__(256)
void cvt_kernel(const float* __restrict__ in, bf16_t* __restrict__ out, int n) {
  int i = blockIdx.x * blockDim.x + threadIdx.x;
  if (i < n) out[i] = (bf16_t)in[i];
}

// ---------------------------------------------------------------------------
// 3) Tiled WMMA GEMM:  C[M,N] = A[M,K] * W[N,K]^T   (A,W bf16)
//    Block tile 128x128, 8 waves (4Mx2N) of 32x64, K-step 32,
//    double-buffered LDS filled with async global->LDS copies.
//    outb != null -> bf16 store; else f32 store with residual.
// ---------------------------------------------------------------------------
#define GT_LDA 40                       // LDS row stride (bf16) for 32-wide tile
#define GT_ABUF (128 * GT_LDA)          // 5120 bf16 per matrix tile
__global__ __launch_bounds__(256)
void gemm_kernel(const bf16_t* __restrict__ A, const bf16_t* __restrict__ W,
                 bf16_t* __restrict__ outb, float* __restrict__ outf,
                 const float* __restrict__ resid, int M, int N, int K) {
  __shared__ __align__(16) bf16_t smem[2 * 2 * GT_ABUF];   // 40 KB
  const int tid  = threadIdx.x;
  const int lane = tid & 31;
  const int wave = tid >> 5;
  const int wm = wave & 3;               // 4 M-slices of 32
  const int wn = wave >> 2;              // 2 N-slices of 64
  const int bm = blockIdx.y, bn = blockIdx.x;
  const int hi = lane >> 4;
  const int ln = lane & 15;

  const int crow  = tid >> 1;            // 0..127 copy row
  const int ccol  = (tid & 1) * 16;      // 0 or 16

  v8f acc[2][4] = {};

  auto copy_tile = [&](int buf, int k0) {
    bf16_t* As = smem + buf * 2 * GT_ABUF;
    bf16_t* Bs = As + GT_ABUF;
    const bf16_t* ga = A + ((size_t)(bm * 128 + crow) * K + k0 + ccol);
    const bf16_t* gb = W + ((size_t)(bn * 128 + crow) * K + k0 + ccol);
    async_copy_b128(&As[crow * GT_LDA + ccol],     ga);
    async_copy_b128(&As[crow * GT_LDA + ccol + 8], ga + 8);
    async_copy_b128(&Bs[crow * GT_LDA + ccol],     gb);
    async_copy_b128(&Bs[crow * GT_LDA + ccol + 8], gb + 8);
  };

  const int nk = K / 32;
  int cur = 0;
  copy_tile(cur, 0);
  wait_async0();
  __syncthreads();

  for (int step = 0; step < nk; ++step) {
    if (step + 1 < nk) copy_tile(cur ^ 1, (step + 1) * 32);

    const bf16_t* As = smem + cur * 2 * GT_ABUF;
    const bf16_t* Bs = As + GT_ABUF;

    v16bf afrag[2], bfrag[4];
    for (int mi = 0; mi < 2; ++mi) {
      const bf16_t* base = &As[(wm * 32 + mi * 16 + ln) * GT_LDA];
      uint4* f = (uint4*)&afrag[mi];
      f[0] = *(const uint4*)(base + (hi ? 8 : 0));
      f[1] = *(const uint4*)(base + (hi ? 24 : 16));
    }
    for (int ni = 0; ni < 4; ++ni) {
      const bf16_t* base = &Bs[(wn * 64 + ni * 16 + ln) * GT_LDA];
      uint4* f = (uint4*)&bfrag[ni];
      f[0] = *(const uint4*)(base + (hi ? 8 : 0));
      f[1] = *(const uint4*)(base + (hi ? 24 : 16));
    }
    for (int mi = 0; mi < 2; ++mi)
      for (int ni = 0; ni < 4; ++ni)
        acc[mi][ni] = wmma_bf16(afrag[mi], bfrag[ni], acc[mi][ni]);

    wait_async0();       // next tile landed
    __syncthreads();     // everyone done reading current tile
    cur ^= 1;
  }

  for (int mi = 0; mi < 2; ++mi)
    for (int ni = 0; ni < 4; ++ni) {
      const int col = bn * 128 + wn * 64 + ni * 16 + ln;
      for (int j = 0; j < 8; ++j) {
        const int row = bm * 128 + wm * 32 + mi * 16 + (hi ? j + 8 : j);
        const size_t idx = (size_t)row * N + col;
        const float v = acc[mi][ni][j];
        if (outb) outb[idx] = (bf16_t)v;
        else      outf[idx] = resid[idx] + v;
      }
    }
}

// ---------------------------------------------------------------------------
// 4) Flash attention, one wave per (b, head, 16-row q tile).
// ---------------------------------------------------------------------------
__global__ __launch_bounds__(32)
void attn_kernel(const bf16_t* __restrict__ q, const bf16_t* __restrict__ k,
                 const bf16_t* __restrict__ v, bf16_t* __restrict__ o) {
  __shared__ __align__(16) bf16_t Ps[16 * 32];
  __shared__ __align__(16) bf16_t Vs[32 * 72];

  const int lane = threadIdx.x;
  const int hi = lane >> 4;
  const int ln = lane & 15;
  const int qt   = blockIdx.x;
  const int head = blockIdx.y;
  const int b    = blockIdx.z;
  const int g    = head / (H_ / G_);

  v16bf qf[2];
  {
    const bf16_t* qbase = q + ((size_t)(b * S_ + qt * 16 + ln) * D_ + head * DH_);
    for (int kd = 0; kd < 2; ++kd) {
      uint4* f = (uint4*)&qf[kd];
      f[0] = *(const uint4*)(qbase + kd * 32 + (hi ? 8 : 0));
      f[1] = *(const uint4*)(qbase + kd * 32 + (hi ? 24 : 16));
    }
  }

  float mrow[8], lrow[8];
  v8f oacc[4] = {};
  for (int j = 0; j < 8; ++j) { mrow[j] = -1e30f; lrow[j] = 0.f; }

  const bf16_t* kb = k + (size_t)(b * S_) * KV_ + g * DH_;
  const bf16_t* vb = v + (size_t)(b * S_) * KV_ + g * DH_;

  for (int t0 = 0; t0 < S_; t0 += 32) {
    // async-stage V chunk 32x64 (one row per lane), overlapped with QK^T
    {
      const bf16_t* gsrc = vb + (size_t)(t0 + lane) * KV_;
      bf16_t* dst = &Vs[lane * 72];
      for (int i = 0; i < 8; ++i) async_copy_b128(dst + i * 8, gsrc + i * 8);
    }

    v8f sacc[2] = {};
    for (int nt = 0; nt < 2; ++nt)
      for (int kd = 0; kd < 2; ++kd) {
        v16bf kf;
        const uint4* kr = (const uint4*)(kb + (size_t)(t0 + nt * 16 + ln) * KV_
                                         + kd * 32 + (hi ? 8 : 0));
        uint4* f = (uint4*)&kf;
        f[0] = kr[0]; f[1] = kr[2];
        sacc[nt] = wmma_bf16(qf[kd], kf, sacc[nt]);
      }
    wait_async0();
    __syncthreads();

    float escale[8];
    for (int j = 0; j < 8; ++j) {
      float s0 = sacc[0][j] * 0.125f;              // 1/sqrt(dh)
      float s1 = sacc[1][j] * 0.125f;
      float rm = fmaxf(s0, s1);
      for (int msk = 1; msk < 16; msk <<= 1) rm = fmaxf(rm, __shfl_xor(rm, msk, 32));
      const float mnew = fmaxf(mrow[j], rm);
      const float sc = __expf(mrow[j] - mnew);
      const float p0 = __expf(s0 - mnew);
      const float p1 = __expf(s1 - mnew);
      float rs = p0 + p1;
      for (int msk = 1; msk < 16; msk <<= 1) rs += __shfl_xor(rs, msk, 32);
      lrow[j] = lrow[j] * sc + rs;
      mrow[j] = mnew;
      escale[j] = sc;
      const int prow = hi ? j + 8 : j;
      Ps[prow * 32 + ln]      = (bf16_t)p0;
      Ps[prow * 32 + 16 + ln] = (bf16_t)p1;
    }
    for (int nd = 0; nd < 4; ++nd)
      for (int j = 0; j < 8; ++j) oacc[nd][j] *= escale[j];
    __syncthreads();

    v16bf pf;
    {
      const bf16_t* pr = &Ps[ln * 32];
      uint4* f = (uint4*)&pf;
      f[0] = *(const uint4*)(pr + (hi ? 8 : 0));
      f[1] = *(const uint4*)(pr + (hi ? 24 : 16));
    }
    for (int nd = 0; nd < 4; ++nd) {
      v16bf vf;
      for (int i = 0; i < 16; ++i)
        vf[i] = Vs[((i < 8) ? (hi * 8 + i) : (16 + hi * 8 + i - 8)) * 72 + nd * 16 + ln];
      oacc[nd] = wmma_bf16(pf, vf, oacc[nd]);
    }
    __syncthreads();
  }

  bf16_t* ob = o + ((size_t)(b * S_ + qt * 16) * D_ + head * DH_);
  for (int nd = 0; nd < 4; ++nd)
    for (int j = 0; j < 8; ++j) {
      const int row = hi ? j + 8 : j;
      ob[(size_t)row * D_ + nd * 16 + ln] = (bf16_t)(oacc[nd][j] / lrow[j]);
    }
}

// ---------------------------------------------------------------------------
// launch
// ---------------------------------------------------------------------------
extern "C" void kernel_launch(void* const* d_in, const int* in_sizes, int n_in,
                              void* d_out, int out_size, void* d_ws, size_t ws_size,
                              hipStream_t stream) {
  (void)in_sizes; (void)n_in; (void)out_size; (void)ws_size;
  const float* x      = (const float*)d_in[0];
  const float* nscale = (const float*)d_in[1];
  const float* sscale = (const float*)d_in[2];
  const float* Wq     = (const float*)d_in[3];
  const float* Wk     = (const float*)d_in[4];
  const float* Wv     = (const float*)d_in[5];
  const float* Wo     = (const float*)d_in[6];
  float* out = (float*)d_out;

  char* ws = (char*)d_ws;
  size_t off = 0;
  auto alloc = [&](size_t bytes) {
    void* p = ws + off; off += (bytes + 255) & ~(size_t)255; return p;
  };
  bf16_t* h   = (bf16_t*)alloc((size_t)B_ * S_ * D_ * 2);
  bf16_t* wqb = (bf16_t*)alloc((size_t)D_ * D_ * 2);
  bf16_t* wkb = (bf16_t*)alloc((size_t)KV_ * D_ * 2);
  bf16_t* wvb = (bf16_t*)alloc((size_t)KV_ * D_ * 2);
  bf16_t* wob = (bf16_t*)alloc((size_t)D_ * D_ * 2);
  bf16_t* qb  = (bf16_t*)alloc((size_t)B_ * S_ * D_ * 2);
  bf16_t* kbf = (bf16_t*)alloc((size_t)B_ * S_ * KV_ * 2);
  bf16_t* vbf = (bf16_t*)alloc((size_t)B_ * S_ * KV_ * 2);
  bf16_t* ab  = (bf16_t*)alloc((size_t)B_ * S_ * D_ * 2);

  rmsnorm_kernel<<<B_ * S_, 256, 0, stream>>>(x, nscale, sscale, h);

  const int nW = D_ * D_;
  const int nKW = KV_ * D_;
  cvt_kernel<<<(nW  + 255) / 256, 256, 0, stream>>>(Wq, wqb, nW);
  cvt_kernel<<<(nKW + 255) / 256, 256, 0, stream>>>(Wk, wkb, nKW);
  cvt_kernel<<<(nKW + 255) / 256, 256, 0, stream>>>(Wv, wvb, nKW);
  cvt_kernel<<<(nW  + 255) / 256, 256, 0, stream>>>(Wo, wob, nW);

  dim3 gq(D_ / 128, (B_ * S_) / 128);
  dim3 gk(KV_ / 128, (B_ * S_) / 128);
  gemm_kernel<<<gq, 256, 0, stream>>>(h, wqb, qb,  nullptr, nullptr, B_ * S_, D_,  D_);
  gemm_kernel<<<gk, 256, 0, stream>>>(h, wkb, kbf, nullptr, nullptr, B_ * S_, KV_, D_);
  gemm_kernel<<<gk, 256, 0, stream>>>(h, wvb, vbf, nullptr, nullptr, B_ * S_, KV_, D_);

  dim3 ga(S_ / 16, H_, B_);
  attn_kernel<<<ga, 32, 0, stream>>>(qb, kbf, vbf, ab);

  gemm_kernel<<<gq, 256, 0, stream>>>(ab, wob, nullptr, out, x, B_ * S_, D_, D_);
}
